// Decoder_1D_Matryoshka_73186242723923
// MI455X (gfx1250) — compile-verified, matrix operand order checked by reading, not run on hardware
//
#include <hip/hip_runtime.h>
#include <hip/hip_bf16.h>

typedef __bf16 bf16_t;
typedef __attribute__((ext_vector_type(16))) __bf16 v16bf;
typedef __attribute__((ext_vector_type(8)))  float  v8f;

#define D_MODEL 1024
#define N_HEAD  16
#define DHEAD   64
#define SEQ     512
#define MTOK    256
#define BATCH   16
#define NLAYER  12
#define ROWS    (BATCH * SEQ)   // 8192

// ---------- helpers ----------

__device__ __forceinline__ bf16_t f2bf(float f) {
  unsigned u = __builtin_bit_cast(unsigned, f);
  unsigned r = (u + 0x7FFFu + ((u >> 16) & 1u)) >> 16;
  unsigned short s = (unsigned short)r;
  return __builtin_bit_cast(bf16_t, s);
}

// Load one 16x32 bf16 fragment (A layout; also used for B since weights are
// stored transposed [N x K]).  ISA 16-bit A layout:
//   lanes 0-15 (row = lane):   VGPR0..3 -> K pairs 0..7, VGPR4..7 -> K 16..23
//   lanes 16-31 (row = lane-16): same + 8
__device__ __forceinline__ v16bf load_frag(const bf16_t* base, int ld) {
  const int lane = threadIdx.x & 31;
  const int r    = lane & 15;
  const int kh   = lane >> 4;
  const bf16_t* p = base + (size_t)r * ld + kh * 8;
  union { v16bf v; unsigned u[8]; } f;
#pragma unroll
  for (int i = 0; i < 8; ++i) {
    const int ko = (i < 4) ? (2 * i) : (16 + 2 * (i - 4));
    f.u[i] = *(const unsigned*)(p + ko);
  }
  return f.v;
}

__device__ __forceinline__ v8f zero8() {
  v8f z = {0.f, 0.f, 0.f, 0.f, 0.f, 0.f, 0.f, 0.f};
  return z;
}

#define WMMA_BF16(A_, B_, C_) \
  __builtin_amdgcn_wmma_f32_16x16x32_bf16(false, (A_), false, (B_), (short)0, (C_), false, false)

// gfx1250 async global->LDS copy (ASYNCcnt).  GV mode: VDST = LDS byte offset,
// VADDR = 64-bit global address (per CDNA5 ISA).
__device__ __forceinline__ void async_copy_b128(unsigned lds_off, const void* gsrc) {
  asm volatile("global_load_async_to_lds_b128 %0, %1, off"
               :: "v"(lds_off), "v"(gsrc)
               : "memory");
}
__device__ __forceinline__ void wait_async0() {
  asm volatile("s_wait_asynccnt 0x0" ::: "memory");
}

// Generic 16x16 single-tile GEMM (used only for the tiny final projection).
__device__ __forceinline__ v8f gemm_tile(const bf16_t* A, int lda,
                                         const bf16_t* BT, int ldb, int Kdim) {
  v8f acc = zero8();
  for (int k0 = 0; k0 < Kdim; k0 += 32) {
    v16bf af = load_frag(A + k0, lda);
    v16bf bf = load_frag(BT + k0, ldb);
    acc = WMMA_BF16(af, bf, acc);
  }
  return acc;
}

// ---------- block GEMM core: 64(M) x 128(N) per 256-thread block ----------
// 8 waves = 2 (M) x 4 (N); each wave computes 32x32 via 2x2 WMMA accumulators.
// A tile (64 x 32 per k-step) is staged into LDS with async copies, double
// buffered so the copy for k-step i+1 overlaps compute on k-step i.
struct Acc22 { v8f a[2][2]; };

__device__ __forceinline__ void gemm_block64x128(
    const bf16_t* __restrict__ Ablk,   // global, row (blockIdx.y*64), col 0
    int lda,
    const bf16_t* __restrict__ BTwave, // global, rows tn..tn+31 of B^T [N x K]
    int Kdim, bf16_t* lds /* [2][64*32] */, int wm, Acc22& acc) {
#pragma unroll
  for (int i = 0; i < 2; ++i)
#pragma unroll
    for (int j = 0; j < 2; ++j) acc.a[i][j] = zero8();

  const int t = threadIdx.x;
  const int srow = t >> 2, schunk = t & 3;      // 64 rows x 4 x 16B = 4KB tile
  const char* gsrc = (const char*)(Ablk + (size_t)srow * lda + schunk * 8);
  const unsigned lbase = (unsigned)(uintptr_t)lds + (unsigned)(srow * 64 + schunk * 16);

  async_copy_b128(lbase, gsrc);                 // preload k0 = 0 into buffer 0
  wait_async0();
  __syncthreads();

  for (int k0 = 0; k0 < Kdim; k0 += 32) {
    const int p = (k0 >> 5) & 1;
    if (k0 + 32 < Kdim)                         // stage next tile into other buf
      async_copy_b128(lbase + (unsigned)((p ^ 1) * 4096), gsrc + (size_t)(k0 + 32) * 2);

    const bf16_t* abase = lds + p * 2048 + (wm * 32) * 32;
    v16bf a0 = load_frag(abase, 32);
    v16bf a1 = load_frag(abase + 16 * 32, 32);
    v16bf b0 = load_frag(BTwave + k0, Kdim);
    v16bf b1 = load_frag(BTwave + (size_t)16 * Kdim + k0, Kdim);
    acc.a[0][0] = WMMA_BF16(a0, b0, acc.a[0][0]);
    acc.a[0][1] = WMMA_BF16(a0, b1, acc.a[0][1]);
    acc.a[1][0] = WMMA_BF16(a1, b0, acc.a[1][0]);
    acc.a[1][1] = WMMA_BF16(a1, b1, acc.a[1][1]);

    wait_async0();
    __syncthreads();
  }
}

__device__ __forceinline__ void wave_coords(int& wm, int& wn, int& n, int& kh) {
  int wid = threadIdx.x >> 5;
  wm = wid >> 2;            // 0..1
  wn = wid & 3;             // 0..3
  n  = threadIdx.x & 15;
  kh = (threadIdx.x >> 4) & 1;
}

// ---------- embedding ----------

__global__ void embed_kernel(const float* __restrict__ latents,
                             const float* __restrict__ input_w,
                             const float* __restrict__ input_b,
                             const float* __restrict__ pos_full,
                             const float* __restrict__ lat_pos,
                             const float* __restrict__ mask_tok,
                             float* __restrict__ x) {
  int row = blockIdx.x;                // b*512 + l
  int b = row >> 9, l = row & 511;
  int t = threadIdx.x;
#pragma unroll
  for (int j = 0; j < 4; ++j) {
    int d = j * 256 + t;
    float val;
    if (l < MTOK) {
      val = mask_tok[d] + pos_full[(size_t)l * D_MODEL + d];
    } else {
      int li = l - MTOK;
      val = input_b[d] + lat_pos[(size_t)li * D_MODEL + d];
      const float* lr = latents + (size_t)(b * 256 + li) * 32;
#pragma unroll
      for (int i = 0; i < 32; ++i) val += lr[i] * input_w[(size_t)i * D_MODEL + d];
    }
    x[(size_t)row * D_MODEL + d] = val;
  }
}

// ---------- layernorm (block per row) ----------

__global__ void ln_kernel(const float* __restrict__ in, const float* __restrict__ g,
                          const float* __restrict__ bb, float* __restrict__ outf,
                          bf16_t* __restrict__ outb, int rows_per_b, int b_stride) {
  int row = blockIdx.x;
  int ir = (row / rows_per_b) * b_stride + (row % rows_per_b);
  const float* xi = in + (size_t)ir * D_MODEL;
  int t = threadIdx.x;
  __shared__ float red[256];
  float v[4];
  float s = 0.f;
#pragma unroll
  for (int j = 0; j < 4; ++j) { v[j] = xi[j * 256 + t]; s += v[j]; }
  red[t] = s; __syncthreads();
  for (int o = 128; o > 0; o >>= 1) { if (t < o) red[t] += red[t + o]; __syncthreads(); }
  float mu = red[0] * (1.f / 1024.f);
  __syncthreads();
  float s2 = 0.f;
#pragma unroll
  for (int j = 0; j < 4; ++j) { float d = v[j] - mu; s2 += d * d; }
  red[t] = s2; __syncthreads();
  for (int o = 128; o > 0; o >>= 1) { if (t < o) red[t] += red[t + o]; __syncthreads(); }
  float rstd = rsqrtf(red[0] * (1.f / 1024.f) + 1e-5f);
#pragma unroll
  for (int j = 0; j < 4; ++j) {
    int d = j * 256 + t;
    float y = (v[j] - mu) * rstd * g[d] + bb[d];
    if (outf) outf[(size_t)row * D_MODEL + d] = y;
    if (outb) outb[(size_t)row * D_MODEL + d] = f2bf(y);
  }
}

// ---------- weight convert + transpose: f32 [K x N] -> bf16 [N x K] ----------

__global__ void convT_kernel(const float* __restrict__ src, bf16_t* __restrict__ dst,
                             int Kd, int Nd) {
  int idx = blockIdx.x * 256 + threadIdx.x;
  if (idx >= Kd * Nd) return;
  int nn = idx % Nd, kk = idx / Nd;
  dst[(size_t)nn * Kd + kk] = f2bf(src[idx]);
}

// ---------- GEMM kernels (block 64x128, wave 32x32) ----------

// QKV: [8192x1024]x[1024x3072] + bias, scatter to q[B,H,L,DH], k[B,H,L,DH], vT[B,H,DH,L]
__global__ void gemm_qkv_kernel(const bf16_t* __restrict__ A, const bf16_t* __restrict__ BT,
                                const float* __restrict__ bias,
                                bf16_t* __restrict__ q, bf16_t* __restrict__ k,
                                bf16_t* __restrict__ vT) {
  __shared__ __align__(16) bf16_t atile[2][64 * 32];
  int wm, wn, n, kh; wave_coords(wm, wn, n, kh);
  int tmb = blockIdx.y * 64;
  int tnw = blockIdx.x * 128 + wn * 32;
  Acc22 acc;
  gemm_block64x128(A + (size_t)tmb * D_MODEL, D_MODEL,
                   BT + (size_t)tnw * D_MODEL, D_MODEL, &atile[0][0], wm, acc);
#pragma unroll
  for (int sn = 0; sn < 2; ++sn) {
    int gc = tnw + sn * 16 + n;
    float bv = bias[gc];
    int sec = gc >> 10, cc = gc & 1023;
    int head = cc >> 6, dh = cc & 63;
#pragma unroll
    for (int sm = 0; sm < 2; ++sm) {
#pragma unroll
      for (int r = 0; r < 8; ++r) {
        int gr = tmb + wm * 32 + sm * 16 + r + 8 * kh;
        int b = gr >> 9, l = gr & 511;
        bf16_t val = f2bf(acc.a[sm][sn][r] + bv);
        size_t bh = (size_t)(b * N_HEAD + head);
        if (sec == 0)      q[(bh * SEQ + l) * DHEAD + dh] = val;
        else if (sec == 1) k[(bh * SEQ + l) * DHEAD + dh] = val;
        else               vT[(bh * DHEAD + dh) * SEQ + l] = val;
      }
    }
  }
}

// C = A*B^T + bias, accumulated into f32 residual stream x (x stride 1024).
__global__ void gemm_add_kernel(const bf16_t* __restrict__ A, const bf16_t* __restrict__ BT,
                                const float* __restrict__ bias, float* __restrict__ x,
                                int Kdim) {
  __shared__ __align__(16) bf16_t atile[2][64 * 32];
  int wm, wn, n, kh; wave_coords(wm, wn, n, kh);
  int tmb = blockIdx.y * 64;
  int tnw = blockIdx.x * 128 + wn * 32;
  Acc22 acc;
  gemm_block64x128(A + (size_t)tmb * Kdim, Kdim,
                   BT + (size_t)tnw * Kdim, Kdim, &atile[0][0], wm, acc);
#pragma unroll
  for (int sn = 0; sn < 2; ++sn) {
    int gc = tnw + sn * 16 + n;
    float bv = bias[gc];
#pragma unroll
    for (int sm = 0; sm < 2; ++sm) {
#pragma unroll
      for (int r = 0; r < 8; ++r) {
        int gr = tmb + wm * 32 + sm * 16 + r + 8 * kh;
        x[(size_t)gr * D_MODEL + gc] += acc.a[sm][sn][r] + bv;
      }
    }
  }
}

// C = gelu(A*B^T + bias) -> bf16 (out stride Ntot)
__global__ void gemm_gelu_kernel(const bf16_t* __restrict__ A, const bf16_t* __restrict__ BT,
                                 const float* __restrict__ bias, bf16_t* __restrict__ out,
                                 int Kdim, int Ntot) {
  __shared__ __align__(16) bf16_t atile[2][64 * 32];
  int wm, wn, n, kh; wave_coords(wm, wn, n, kh);
  int tmb = blockIdx.y * 64;
  int tnw = blockIdx.x * 128 + wn * 32;
  Acc22 acc;
  gemm_block64x128(A + (size_t)tmb * Kdim, Kdim,
                   BT + (size_t)tnw * Kdim, Kdim, &atile[0][0], wm, acc);
#pragma unroll
  for (int sn = 0; sn < 2; ++sn) {
    int gc = tnw + sn * 16 + n;
    float bv = bias[gc];
#pragma unroll
    for (int sm = 0; sm < 2; ++sm) {
#pragma unroll
      for (int r = 0; r < 8; ++r) {
        int gr = tmb + wm * 32 + sm * 16 + r + 8 * kh;
        float v = acc.a[sm][sn][r] + bv;
        out[(size_t)gr * Ntot + gc] = f2bf(0.5f * v * (1.f + erff(v * 0.70710678118f)));
      }
    }
  }
}

// Final projection: [4096x1024]x[1024x16] + out_b -> f32 d_out.  2 waves/block.
__global__ void gemm_out_kernel(const bf16_t* __restrict__ A, const bf16_t* __restrict__ BT,
                                const float* __restrict__ bias, float* __restrict__ out) {
  int wid = threadIdx.x >> 5;
  int tm = blockIdx.x * 32 + wid * 16;
  int lane = threadIdx.x & 31;
  int n = lane & 15, kh = lane >> 4;
  v8f acc = gemm_tile(A + (size_t)tm * D_MODEL, D_MODEL, BT, D_MODEL, D_MODEL);
  float bv = bias[n];
#pragma unroll
  for (int r = 0; r < 8; ++r) {
    int gr = tm + r + 8 * kh;
    out[(size_t)gr * 16 + n] = acc[r] + bv;
  }
}

// ---------- attention: one wave per (b, h, 16-row tile) ----------

__global__ void attn_kernel(const bf16_t* __restrict__ q, const bf16_t* __restrict__ kk,
                            const bf16_t* __restrict__ vT, const int* __restrict__ nact,
                            bf16_t* __restrict__ o, float scale) {
  __shared__ float  sc[16][516];   // padded: stride % 64 == 4 banks
  __shared__ bf16_t pr[16][520];
  int rt = blockIdx.x, h = blockIdx.y, b = blockIdx.z;
  int lane = threadIdx.x & 31;
  int n = lane & 15, kh = lane >> 4;
  size_t bh = (size_t)(b * N_HEAD + h);
  const bf16_t* qb = q + (bh * SEQ + rt * 16) * DHEAD;
  v16bf a0 = load_frag(qb, DHEAD);        // K = 0..31
  v16bf a1 = load_frag(qb + 32, DHEAD);   // K = 32..63
  int ni = nact[b];

  for (int ct = 0; ct < 32; ++ct) {
    const bf16_t* kb = kk + (bh * SEQ + ct * 16) * DHEAD;
    v16bf b0 = load_frag(kb, DHEAD);
    v16bf b1 = load_frag(kb + 32, DHEAD);
    v8f c = zero8();
    c = WMMA_BF16(a0, b0, c);
    c = WMMA_BF16(a1, b1, c);
#pragma unroll
    for (int r = 0; r < 8; ++r) {
      int m = r + 8 * kh;
      int gr = rt * 16 + m, gc = ct * 16 + n;
      bool ok = (gr < MTOK && gc < MTOK)
             || (gr < MTOK && gc >= MTOK && gc < MTOK + ni)
             || (gr >= MTOK && gr < MTOK + ni && gc >= MTOK && gc <= gr)
             || (gr >= MTOK + ni && gc >= MTOK && gc < MTOK + ni);
      sc[m][gc] = ok ? c[r] * scale : -3.0e38f;
    }
  }
  __syncthreads();

  if (lane < 16) {   // softmax, one lane per row
    float mx = -3.4e38f;
    for (int c = 0; c < SEQ; ++c) mx = fmaxf(mx, sc[lane][c]);
    float sm = 0.f;
    for (int c = 0; c < SEQ; ++c) sm += __expf(sc[lane][c] - mx);
    float inv = 1.f / sm;
    for (int c = 0; c < SEQ; ++c) pr[lane][c] = f2bf(__expf(sc[lane][c] - mx) * inv);
  }
  __syncthreads();

  for (int nt = 0; nt < 4; ++nt) {       // o = P(16x512) x V(512x64)
    v8f acc = zero8();
    const bf16_t* vb = vT + (bh * DHEAD + nt * 16) * SEQ;
    for (int kt = 0; kt < 16; ++kt) {
      v16bf af = load_frag(&pr[0][0] + kt * 32, 520);
      v16bf bf = load_frag(vb + kt * 32, SEQ);
      acc = WMMA_BF16(af, bf, acc);
    }
#pragma unroll
    for (int r = 0; r < 8; ++r) {
      int m = r + 8 * kh;
      int gr = b * SEQ + rt * 16 + m;
      o[(size_t)gr * D_MODEL + h * DHEAD + nt * 16 + n] = f2bf(acc[r]);
    }
  }
}

// ---------- host ----------

extern "C" void kernel_launch(void* const* d_in, const int* in_sizes, int n_in,
                              void* d_out, int out_size, void* d_ws, size_t ws_size,
                              hipStream_t stream) {
  const float* latents   = (const float*)d_in[0];
  const int*   nact      = (const int*)  d_in[1];
  const float* input_w   = (const float*)d_in[2];
  const float* input_b   = (const float*)d_in[3];
  const float* pos_full  = (const float*)d_in[4];
  const float* lat_pos   = (const float*)d_in[5];
  const float* mask_tok  = (const float*)d_in[6];
  const float* ln_pre_g  = (const float*)d_in[7];
  const float* ln_pre_b  = (const float*)d_in[8];
  const float* qkv_w     = (const float*)d_in[9];
  const float* qkv_b     = (const float*)d_in[10];
  const float* proj_w    = (const float*)d_in[11];
  const float* proj_b    = (const float*)d_in[12];
  const float* ln1_g     = (const float*)d_in[13];
  const float* ln1_b     = (const float*)d_in[14];
  const float* ln2_g     = (const float*)d_in[15];
  const float* ln2_b     = (const float*)d_in[16];
  const float* fc1_w     = (const float*)d_in[17];
  const float* fc1_b     = (const float*)d_in[18];
  const float* fc2_w     = (const float*)d_in[19];
  const float* fc2_b     = (const float*)d_in[20];
  const float* ln_post_g = (const float*)d_in[21];
  const float* ln_post_b = (const float*)d_in[22];
  const float* out_w     = (const float*)d_in[23];
  const float* out_b     = (const float*)d_in[24];

  char* ws = (char*)d_ws;
  size_t off = 0;
  auto alloc = [&](size_t bytes) -> void* {
    void* p = ws + off;
    off += (bytes + 255) & ~(size_t)255;
    return p;
  };
  float*  x     = (float*) alloc((size_t)ROWS * D_MODEL * 4);
  bf16_t* hbuf  = (bf16_t*)alloc((size_t)ROWS * D_MODEL * 2);
  bf16_t* qb    = (bf16_t*)alloc((size_t)ROWS * D_MODEL * 2);
  bf16_t* kb    = (bf16_t*)alloc((size_t)ROWS * D_MODEL * 2);
  bf16_t* vT    = (bf16_t*)alloc((size_t)ROWS * D_MODEL * 2);
  bf16_t* obuf  = (bf16_t*)alloc((size_t)ROWS * D_MODEL * 2);
  bf16_t* ff    = (bf16_t*)alloc((size_t)ROWS * 4096 * 2);
  bf16_t* qkvT  = (bf16_t*)alloc((size_t)3072 * 1024 * 2);
  bf16_t* projT = (bf16_t*)alloc((size_t)1024 * 1024 * 2);
  bf16_t* fc1T  = (bf16_t*)alloc((size_t)4096 * 1024 * 2);
  bf16_t* fc2T  = (bf16_t*)alloc((size_t)1024 * 4096 * 2);
  bf16_t* outT  = (bf16_t*)alloc((size_t)16 * 1024 * 2);

  dim3 b256(256);
  embed_kernel<<<ROWS, b256, 0, stream>>>(latents, input_w, input_b, pos_full,
                                          lat_pos, mask_tok, x);
  ln_kernel<<<ROWS, b256, 0, stream>>>(x, ln_pre_g, ln_pre_b, x, nullptr, SEQ, SEQ);

  const float scale = 0.125f;   // DH^-0.5
  for (int l = 0; l < NLAYER; ++l) {
    convT_kernel<<<(1024 * 3072 + 255) / 256, b256, 0, stream>>>(
        qkv_w + (size_t)l * 1024 * 3072, qkvT, 1024, 3072);
    convT_kernel<<<(1024 * 1024 + 255) / 256, b256, 0, stream>>>(
        proj_w + (size_t)l * 1024 * 1024, projT, 1024, 1024);
    convT_kernel<<<(1024 * 4096 + 255) / 256, b256, 0, stream>>>(
        fc1_w + (size_t)l * 1024 * 4096, fc1T, 1024, 4096);
    convT_kernel<<<(4096 * 1024 + 255) / 256, b256, 0, stream>>>(
        fc2_w + (size_t)l * 4096 * 1024, fc2T, 4096, 1024);

    ln_kernel<<<ROWS, b256, 0, stream>>>(x, ln1_g + l * 1024, ln1_b + l * 1024,
                                         nullptr, hbuf, SEQ, SEQ);
    gemm_qkv_kernel<<<dim3(24, 128), b256, 0, stream>>>(hbuf, qkvT,
                                                        qkv_b + l * 3072, qb, kb, vT);
    attn_kernel<<<dim3(32, 16, 16), dim3(32), 0, stream>>>(qb, kb, vT, nact, obuf, scale);
    gemm_add_kernel<<<dim3(8, 128), b256, 0, stream>>>(obuf, projT,
                                                       proj_b + l * 1024, x, 1024);
    ln_kernel<<<ROWS, b256, 0, stream>>>(x, ln2_g + l * 1024, ln2_b + l * 1024,
                                         nullptr, hbuf, SEQ, SEQ);
    gemm_gelu_kernel<<<dim3(32, 128), b256, 0, stream>>>(hbuf, fc1T,
                                                         fc1_b + l * 4096, ff, 1024, 4096);
    gemm_add_kernel<<<dim3(8, 128), b256, 0, stream>>>(ff, fc2T,
                                                       fc2_b + l * 1024, x, 4096);
  }

  convT_kernel<<<(1024 * 16 + 255) / 256, b256, 0, stream>>>(out_w, outT, 1024, 16);
  ln_kernel<<<BATCH * MTOK, b256, 0, stream>>>(x, ln_post_g, ln_post_b,
                                               nullptr, hbuf, MTOK, SEQ);
  gemm_out_kernel<<<dim3(128), dim3(64), 0, stream>>>(hbuf, outT, out_b, (float*)d_out);
}
